// ClockworkRNN_64991445123176
// MI455X (gfx1250) — matrix operand order repaired and checked by reading, not compiled, not run on hardware
//
#include <hip/hip_runtime.h>
#include <hip/hip_bf16.h>

typedef _Float16 f16_t;
typedef __attribute__((ext_vector_type(16))) _Float16 v16h;
typedef __attribute__((ext_vector_type(8)))  float    v8f;

#define NI     128
#define NO     128
#define KDIM   512
#define TSTEPS 512
#define BATCH  256
#define BTILE  16
#define NWAVES 8

#define HS_STRIDE 520   // f16, row = 1040B = 65*16 (aligned, bank-staggered)
#define HF_STRIDE 520   // f32 carry state
#define XS_STRIDE 136   // f16, row = 272B = 17*16

// workspace layout (bytes)
#define OFF_WHB 0
#define OFF_WIB (OFF_WHB + 512*512*2)
#define OFF_WOB (OFF_WIB + 512*128*2)
#define OFF_BH  (OFF_WOB + 128*512*2)
#define OFF_BO  (OFF_BH  + 512*4)

// ---------- weight prep: f32 -> f16, fuse biases ----------
__global__ void cwrnn_prep(const float* __restrict__ Wi,
                           const float* __restrict__ Wh,
                           const float* __restrict__ Wo,
                           f16_t* __restrict__ Whb, f16_t* __restrict__ Wib,
                           f16_t* __restrict__ Wob, float* __restrict__ bH,
                           float* __restrict__ bO) {
  const int i = blockIdx.x * blockDim.x + threadIdx.x;
  const int stride = gridDim.x * blockDim.x;
  for (int e = i; e < 512 * 512; e += stride)
    Whb[e] = (f16_t)Wh[(e >> 9) * 513 + (e & 511)];
  for (int e = i; e < 512 * 128; e += stride)
    Wib[e] = (f16_t)Wi[(e >> 7) * 129 + (e & 127)];
  for (int e = i; e < 128 * 512; e += stride)
    Wob[e] = (f16_t)Wo[(e >> 9) * 513 + (e & 511)];
  for (int e = i; e < 512; e += stride)
    bH[e] = Wi[e * 129 + 128] + Wh[e * 513 + 512];
  for (int e = i; e < 128; e += stride)
    bO[e] = Wo[e * 513 + 512];
}

// ---------- WMMA fragment loaders ----------
// A (16x32, f16, row-major W): lanes 0-15 hold M=lane, K ktile*32+{0..7,16..23};
// lanes 16-31 same M rows, K +8 / +24 (ISA 16-bit A layout).
__device__ __forceinline__ v16h load_a_global(const f16_t* __restrict__ W,
                                              int kdim, int Mb, int kbase,
                                              int nIdx, int mHalf) {
  const f16_t* p = W + (size_t)(Mb + nIdx) * kdim + kbase + mHalf;
  v16h a;
  ((uint4*)&a)[0] = *(const uint4*)p;
  ((uint4*)&a)[1] = *(const uint4*)(p + 16);
  return a;
}

// B (32x16, f16): lanes 0-15 hold N=lane, K ktile*32+{0..7,16..23};
// lanes 16-31 hold K +8 / +24 (matches A layout). LDS stored as [n][k].
__device__ __forceinline__ v16h load_b_lds(const f16_t* S, int stride,
                                           int kbase, int nIdx, int kHalf) {
  const f16_t* p = S + nIdx * stride + kbase + (kHalf >> 1);
  v16h b;
  ((uint4*)&b)[0] = *(const uint4*)p;
  ((uint4*)&b)[1] = *(const uint4*)(p + 16);
  return b;
}

// Fully-unrolled Wh accumulation tail starting at compile-time ktile KT0.
template <int KT0>
__device__ __forceinline__ v8f wh_accum(const f16_t* __restrict__ Whb,
                                        const f16_t* Hs, int Mb, int nIdx,
                                        int mHalf, int kHalf, v8f acc) {
#pragma unroll
  for (int kt = KT0; kt < 16; ++kt) {
    v16h a = load_a_global(Whb, KDIM, Mb, kt * 32, nIdx, mHalf);
    v16h b = load_b_lds(Hs, HS_STRIDE, kt * 32, nIdx, kHalf);
    acc = __builtin_amdgcn_wmma_f32_16x16x32_f16(false, a, false, b, (short)0,
                                                 acc, false, false);
  }
  return acc;
}

__device__ __forceinline__ v8f wi_accum(const f16_t* __restrict__ Wib,
                                        const f16_t* xs, int Mb, int nIdx,
                                        int mHalf, int kHalf, v8f acc) {
#pragma unroll
  for (int kt = 0; kt < 4; ++kt) {
    v16h a = load_a_global(Wib, NI, Mb, kt * 32, nIdx, mHalf);
    v16h b = load_b_lds(xs, XS_STRIDE, kt * 32, nIdx, kHalf);
    acc = __builtin_amdgcn_wmma_f32_16x16x32_f16(false, a, false, b, (short)0,
                                                 acc, false, false);
  }
  return acc;
}

// ---------- persistent recurrence kernel: one WG owns 16 batch columns ----------
__global__ __launch_bounds__(256, 1) void cwrnn_main(
    const float* __restrict__ X, const f16_t* __restrict__ Whb,
    const f16_t* __restrict__ Wib, const f16_t* __restrict__ Wob,
    const float* __restrict__ biasH, const float* __restrict__ biasO,
    float* __restrict__ outY, float* __restrict__ outH) {
  __shared__ f16_t Hs[BTILE * HS_STRIDE];  // f16 state for WMMA
  __shared__ float Hf[BTILE * HF_STRIDE];  // f32 carried state
  __shared__ f16_t xs[BTILE * XS_STRIDE];  // staged x_t, [b][i]
  __shared__ float bHs[KDIM];
  __shared__ float bOs[NO];

  const int tid  = threadIdx.x;
  const int lane = tid & 31;
  // readfirstlane -> SGPR: tile branches are scalar, EXEC stays all-1s (WMMA req).
  const int wave  = __builtin_amdgcn_readfirstlane(tid >> 5);
  const int waveHi = __builtin_amdgcn_readfirstlane((tid >> 5) >> 2);  // wave>>2
  const int b0    = blockIdx.x * BTILE;
  const int nIdx  = lane & 15;
  const int mHalf = (lane & 16) ? 8 : 0;   // A-frag K sub-offset / C-frag M offset
  const int kHalf = (lane & 16) ? 16 : 0;  // B-frag K sub-offset

  for (int e = tid; e < BTILE * HS_STRIDE; e += 256) Hs[e] = (f16_t)0.0f;
  for (int e = tid; e < BTILE * HF_STRIDE; e += 256) Hf[e] = 0.0f;
  for (int e = tid; e < KDIM; e += 256) bHs[e] = biasH[e];
  if (tid < NO) bOs[tid] = biasO[tid];
  // stage x_0
  for (int e = tid; e < BTILE * NI; e += 256) {
    const int bb = e >> 7, ii = e & (NI - 1);
    xs[bb * XS_STRIDE + ii] =
        (f16_t)__builtin_nontemporal_load(&X[(size_t)(b0 + bb) * NI + ii]);
  }
  __syncthreads();

  for (int t = 0; t < TSTEPS; ++t) {
    // active row blocks at step t: 0..min(7, ctz(t))  (t==0 -> all 8)
    int z = t ? __builtin_ctz(t) : 7;
    if (z > 7) z = 7;
    const int nTiles = 4 * (z + 1);  // active 16-row M tiles

    // ---- phase 1a: Z = Wh@H + Wi@x for this wave's active tiles ----
    // tile = wave + 8j  =>  clock block mb = 2j + (wave>>2): only two
    // possible Wh start points per j -> two fully unrolled tails.
    v8f accs[4];
#define PHASE1_TILE(J, KT0A, KT0B)                                            \
  if (wave + (J) * NWAVES < nTiles) {                                         \
    const int Mb = (wave + (J) * NWAVES) * 16;                                \
    v8f acc = {};                                                             \
    if (waveHi == 0)                                                          \
      acc = wh_accum<KT0A>(Whb, Hs, Mb, nIdx, mHalf, kHalf, acc);             \
    else                                                                      \
      acc = wh_accum<KT0B>(Whb, Hs, Mb, nIdx, mHalf, kHalf, acc);             \
    acc = wi_accum(Wib, xs, Mb, nIdx, mHalf, kHalf, acc);                     \
    accs[J] = acc;                                                            \
  }
    PHASE1_TILE(0, 0, 2)
    PHASE1_TILE(1, 4, 6)
    PHASE1_TILE(2, 8, 10)
    PHASE1_TILE(3, 12, 14)
#undef PHASE1_TILE
    __syncthreads();  // all reads of Hs done before updates

    // ---- phase 1b: write updated H (inactive rows carry H_prev) ----
#pragma unroll
    for (int j = 0; j < 4; ++j) {
      if (wave + j * NWAVES < nTiles) {
        const int Mb = (wave + j * NWAVES) * 16;
#pragma unroll
        for (int r = 0; r < 8; ++r) {
          const int m = Mb + mHalf + r;  // C layout: VGPR r -> M = r + 8*lane[4]
          const float h = tanhf(accs[j][r] + bHs[m]);
          Hf[nIdx * HF_STRIDE + m] = h;
          Hs[nIdx * HS_STRIDE + m] = (f16_t)h;
        }
      }
    }
    __syncthreads();

    // ---- stage x_{t+1} early (hides under Wo WMMAs; read after barrier) ----
    if (t + 1 < TSTEPS) {
      for (int e = tid; e < BTILE * NI; e += 256) {
        const int bb = e >> 7, ii = e & (NI - 1);
        xs[bb * XS_STRIDE + ii] = (f16_t)__builtin_nontemporal_load(
            &X[((size_t)(t + 1) * BATCH + b0 + bb) * NI + ii]);
      }
    }

    // ---- phase 2: y_t = tanh(Wo @ H + bo), one 16-row tile per wave ----
    {
      const int Mb = wave * 16;  // scalar
      v8f acc = {};
#pragma unroll
      for (int kt = 0; kt < 16; ++kt) {
        v16h a = load_a_global(Wob, KDIM, Mb, kt * 32, nIdx, mHalf);
        v16h b = load_b_lds(Hs, HS_STRIDE, kt * 32, nIdx, kHalf);
        acc = __builtin_amdgcn_wmma_f32_16x16x32_f16(
            false, a, false, b, (short)0, acc, false, false);
      }
      float* yout = outY + ((size_t)t * BATCH + b0 + nIdx) * NO + Mb + mHalf;
#pragma unroll
      for (int r = 0; r < 8; ++r)
        __builtin_nontemporal_store(tanhf(acc[r] + bOs[Mb + mHalf + r]),
                                    &yout[r]);
    }
    __syncthreads();  // Hs/xs consistent before next step
  }

  // final H (K, B) from f32 carry state
  for (int e = tid; e < KDIM * BTILE; e += 256) {
    const int k = e >> 4, bb = e & 15;
    outH[(size_t)k * BATCH + b0 + bb] = Hf[bb * HF_STRIDE + k];
  }
}

extern "C" void kernel_launch(void* const* d_in, const int* in_sizes, int n_in,
                              void* d_out, int out_size, void* d_ws,
                              size_t ws_size, hipStream_t stream) {
  (void)in_sizes; (void)n_in; (void)out_size; (void)ws_size;
  const float* X  = (const float*)d_in[0];  // (T, B, NI)
  const float* Wi = (const float*)d_in[1];  // (K, NI+1)
  const float* Wh = (const float*)d_in[2];  // (K, K+1)
  const float* Wo = (const float*)d_in[3];  // (NO, K+1)

  char* ws = (char*)d_ws;
  f16_t* Whb = (f16_t*)(ws + OFF_WHB);
  f16_t* Wib = (f16_t*)(ws + OFF_WIB);
  f16_t* Wob = (f16_t*)(ws + OFF_WOB);
  float* bH  = (float*)(ws + OFF_BH);
  float* bO  = (float*)(ws + OFF_BO);

  float* outY = (float*)d_out;                                   // T*B*NO
  float* outH = (float*)d_out + (size_t)TSTEPS * BATCH * NO;     // K*B

  cwrnn_prep<<<256, 256, 0, stream>>>(Wi, Wh, Wo, Whb, Wib, Wob, bH, bO);
  cwrnn_main<<<BATCH / BTILE, 256, 0, stream>>>(X, Whb, Wib, Wob, bH, bO,
                                                outY, outH);
}
